// NeighborhoodAttention1D_64802466562577
// MI455X (gfx1250) — matrix operand order repaired and checked by reading, not compiled, not run on hardware
//
#include <hip/hip_runtime.h>

// MI455X / gfx1250, wave32. All matrix math via V_WMMA_F32_16X16X32_BF16.

typedef __bf16 bf16;
typedef bf16  v16bf __attribute__((ext_vector_type(16)));
typedef bf16  v8bf  __attribute__((ext_vector_type(8)));
typedef float v8f   __attribute__((ext_vector_type(8)));

union V16 { v16bf v; v8bf h[2]; };

// Native hardware conversion (RNE): one v_cvt per element (packable).
__device__ __forceinline__ bf16 f2bf(float f) { return (bf16)f; }
__device__ __forceinline__ bf16 bf0() { return (bf16)0.0f; }

// Async global->LDS copy path (gfx1250 GLOBAL_LOAD_ASYNC_TO_LDS_B128,
// tracked by ASYNCcnt). Builtin signature (from hipcc diagnostic):
//   void (...)(v4i addrspace(1)* src, v4i addrspace(3)* lds, imm off, imm cpol)
#if defined(__gfx1250__) && __has_builtin(__builtin_amdgcn_global_load_async_to_lds_b128)
#define HAVE_ASYNC_LDS 1
typedef int v4i __attribute__((ext_vector_type(4)));
typedef __attribute__((address_space(1))) v4i* as1_v4i;
typedef __attribute__((address_space(3))) v4i* as3_v4i;
#else
#define HAVE_ASYNC_LDS 0
#endif

__device__ __forceinline__ void wait_asynccnt0() {
#if __has_builtin(__builtin_amdgcn_s_wait_asynccnt)
  __builtin_amdgcn_s_wait_asynccnt(0);
#else
  asm volatile("s_wait_asynccnt 0x0" ::: "memory");
#endif
}

// Problem constants
static constexpr int Bc = 2, Lc = 2048, Cc = 512, Hc = 8, Dc = 64;

// ---------------------------------------------------------------------------
// Kernel 1: qkv = x @ w_qkv^T + b_qkv  (M=4096, N=1536, K=512)
// bf16 inputs in LDS, f32 WMMA accumulation. Epilogue scatters to (B,H,L,D).
// Block = 128 threads = 4 waves in a 2x2 grid; each wave owns a 32x32 patch.
// ---------------------------------------------------------------------------
__global__ __launch_bounds__(128) void k_qkv(
    const float* __restrict__ x, const float* __restrict__ w,
    const float* __restrict__ bias,
    bf16* __restrict__ qb, bf16* __restrict__ kb, bf16* __restrict__ vb)
{
  __shared__ alignas(16) bf16 As[64][32];
  __shared__ alignas(16) bf16 Bs[64][32];
  const int tid = threadIdx.x, lane = tid & 31, wv = tid >> 5;
  const int wm = wv >> 1, wn = wv & 1;
  const int m0 = blockIdx.x * 64, n0 = blockIdx.y * 64;
  const int half = lane >> 4, lr = lane & 15;

  v8f acc[2][2] = {};
  for (int kt = 0; kt < 512; kt += 32) {
    // Prefetch next K-tile (128B/row) while this tile is consumed.
    if (kt + 32 < 512) {
      const size_t pro = (size_t)(tid & 63) * 512 + kt + 32 + ((tid >> 6) << 4);
      __builtin_prefetch(x + (size_t)m0 * 512 + pro, 0, 0);
      __builtin_prefetch(w + (size_t)n0 * 512 + pro, 0, 0);
    }
    __syncthreads();
    // 64x32 f32 tiles of A and B -> bf16 LDS (512 float4 each, 4/thread)
    for (int i = 0; i < 4; ++i) {
      int idx = tid + i * 128;
      int row = idx >> 3, c4 = (idx & 7) << 2;
      float4 fa = *(const float4*)(x + (size_t)(m0 + row) * 512 + kt + c4);
      As[row][c4+0]=f2bf(fa.x); As[row][c4+1]=f2bf(fa.y);
      As[row][c4+2]=f2bf(fa.z); As[row][c4+3]=f2bf(fa.w);
      float4 fb = *(const float4*)(w + (size_t)(n0 + row) * 512 + kt + c4);
      Bs[row][c4+0]=f2bf(fb.x); Bs[row][c4+1]=f2bf(fb.y);
      Bs[row][c4+2]=f2bf(fb.z); Bs[row][c4+3]=f2bf(fb.w);
    }
    __syncthreads();
    // A frag: lane row = M, two 8-elem K chunks (ISA 16-bit A 16x32 layout)
    v16bf af[2], bfr[2];
    for (int mt = 0; mt < 2; ++mt) {
      const bf16* p = &As[wm*32 + mt*16 + lr][half*8];
      V16 u; u.h[0] = *(const v8bf*)p; u.h[1] = *(const v8bf*)(p + 16);
      af[mt] = u.v;
    }
    // B frag: lane col = N, two 8-elem K chunks (same K order as A)
    for (int nt = 0; nt < 2; ++nt) {
      const bf16* p = &Bs[wn*32 + nt*16 + lr][half*8];
      V16 u; u.h[0] = *(const v8bf*)p; u.h[1] = *(const v8bf*)(p + 16);
      bfr[nt] = u.v;
    }
    for (int mt = 0; mt < 2; ++mt)
      for (int nt = 0; nt < 2; ++nt)
        acc[mt][nt] = __builtin_amdgcn_wmma_f32_16x16x32_bf16(
            false, af[mt], false, bfr[nt], (short)0, acc[mt][nt], false, false);
  }
  // Epilogue: C layout VGPR r, lane l: M = r + 8*(l/16), N = l%16
  for (int mt = 0; mt < 2; ++mt)
    for (int nt = 0; nt < 2; ++nt) {
      int n = n0 + wn*32 + nt*16 + lr;
      float bi = bias[n];
      int which = n >> 9, c = n & 511, h = c >> 6, d = c & 63;
      bf16* dst = (which == 0) ? qb : ((which == 1) ? kb : vb);
      for (int r = 0; r < 8; ++r) {
        int row = m0 + wm*32 + mt*16 + r + 8*half;
        int b = row >> 11, ll = row & 2047;
        dst[(((size_t)(b*Hc + h)) * Lc + ll) * Dc + d] = f2bf(acc[mt][nt][r] + bi);
      }
    }
}

// ---------------------------------------------------------------------------
// Kernel 2: neighborhood attention. One wave per 16-query tile.
// Windows of 16 consecutive queries span <=48 consecutive keys, so:
//   scores(16x48) = q(16x64) @ k_span^T  -> 3 N-tiles x 2 K-steps WMMA
//   softmax(masked, +rpb) in LDS
//   out(16x64)    = probs(16x64, zero-padded) @ v_span  -> 4 N-tiles WMMA
// ---------------------------------------------------------------------------
__global__ __launch_bounds__(128) void k_natt(
    const bf16* __restrict__ qb, const bf16* __restrict__ kb,
    const bf16* __restrict__ vb, const float* __restrict__ rpb,
    bf16* __restrict__ ao)
{
  __shared__ alignas(16) float sc[4][16][48];
  __shared__ alignas(16) bf16  pb[4][16][64];
  __shared__ alignas(16) bf16  vt[4][64][64];   // v_span transposed (d, j)
  const int tid = threadIdx.x, lane = tid & 31, wv = tid >> 5;
  const int tile = blockIdx.x * 4 + wv;          // 2048 tiles total
  const int t = tile & 127, bh = tile >> 7;
  const int h = bh & 7, b = bh >> 3;
  const int i0 = t * 16;
  const int half = lane >> 4, lr = lane & 15;
  int base = i0 - 16; if (base < 0) base = 0; if (base > Lc - 33) base = Lc - 33;

  const bf16* qp = qb + (size_t)bh * Lc * Dc;
  const bf16* kp = kb + (size_t)bh * Lc * Dc;
  const bf16* vp = vb + (size_t)bh * Lc * Dc;

  // q A-fragments (16x64 over two K=32 steps)
  v16bf qf[2];
  for (int k2 = 0; k2 < 2; ++k2) {
    const bf16* p = qp + (size_t)(i0 + lr) * Dc + k2*32 + half*8;
    V16 u; u.h[0] = *(const v8bf*)p; u.h[1] = *(const v8bf*)(p + 16);
    qf[k2] = u.v;
  }
  // k B-fragments: B[d][n] = k[base + ntile*16 + n][d]
  v16bf kf[3][2];
  for (int nt = 0; nt < 3; ++nt) {
    int key = base + nt*16 + lr; if (key > Lc - 1) key = Lc - 1;
    for (int k2 = 0; k2 < 2; ++k2) {
      const bf16* p = kp + (size_t)key * Dc + k2*32 + half*8;
      V16 u; u.h[0] = *(const v8bf*)p; u.h[1] = *(const v8bf*)(p + 16);
      kf[nt][k2] = u.v;
    }
  }
  const float scale = 0.125f;   // D^-0.5, D=64
  for (int nt = 0; nt < 3; ++nt) {
    v8f a = {};
    for (int k2 = 0; k2 < 2; ++k2)
      a = __builtin_amdgcn_wmma_f32_16x16x32_bf16(
          false, qf[k2], false, kf[nt][k2], (short)0, a, false, false);
    int col = nt*16 + lr;
    int keyg = base + col;
    for (int r = 0; r < 8; ++r) {
      int row = r + 8*half;
      int i = i0 + row;
      int st = i - 16; if (st < 0) st = 0; if (st > Lc - 33) st = Lc - 33;
      float val;
      if (keyg >= st && keyg <= st + 32)
        val = a[r] * scale + rpb[h*65 + (keyg - i + 32)];
      else
        val = -1e30f;
      sc[wv][row][col] = val;
    }
  }
  // Stage v_span transposed into LDS; zero-pad j = 48..63 (A cols there are 0,
  // but LDS garbage could be NaN, so pad explicitly).
  for (int j = lane; j < 48; j += 32) {
    int key = base + j; if (key > Lc - 1) key = Lc - 1;
    const bf16* p = vp + (size_t)key * Dc;
    for (int d = 0; d < 64; ++d) vt[wv][d][j] = p[d];
  }
  for (int d = lane; d < 64; d += 32)
    for (int j = 48; j < 64; ++j) vt[wv][d][j] = bf0();
  __syncthreads();

  // Per-row softmax (each of lanes 0..15 owns one row)
  if (lane < 16) {
    float mx = -1e30f;
    for (int j = 0; j < 48; ++j) mx = fmaxf(mx, sc[wv][lane][j]);
    float s = 0.f;
    for (int j = 0; j < 48; ++j) {
      float e = __expf(sc[wv][lane][j] - mx);
      sc[wv][lane][j] = e; s += e;
    }
    float inv = 1.f / s;
    for (int j = 0; j < 48; ++j)  pb[wv][lane][j] = f2bf(sc[wv][lane][j] * inv);
    for (int j = 48; j < 64; ++j) pb[wv][lane][j] = bf0();
  }
  __syncthreads();

  // AV: probs(16x64) @ v_span(64x64 padded) -> 16x64
  v16bf pf[2];
  for (int k2 = 0; k2 < 2; ++k2) {
    const bf16* p = &pb[wv][lr][k2*32 + half*8];
    V16 u; u.h[0] = *(const v8bf*)p; u.h[1] = *(const v8bf*)(p + 16);
    pf[k2] = u.v;
  }
  v8f o[4] = {};
  for (int dt = 0; dt < 4; ++dt)
    for (int k2 = 0; k2 < 2; ++k2) {
      const bf16* p = &vt[wv][dt*16 + lr][k2*32 + half*8];
      V16 u; u.h[0] = *(const v8bf*)p; u.h[1] = *(const v8bf*)(p + 16);
      o[dt] = __builtin_amdgcn_wmma_f32_16x16x32_bf16(
          false, pf[k2], false, u.v, (short)0, o[dt], false, false);
    }
  for (int dt = 0; dt < 4; ++dt) {
    int d = dt*16 + lr;
    for (int r = 0; r < 8; ++r) {
      int row = r + 8*half;
      ao[((size_t)(b*Lc + i0 + row)) * Cc + h*64 + d] = f2bf(o[dt][r]);
    }
  }
}

// ---------------------------------------------------------------------------
// Kernel 3: out = attn_out @ w_proj^T + b_proj  (M=4096, N=512, K=512), f32 out
// A tile is already bf16 -> async DMA straight into LDS (ASYNCcnt path).
// ---------------------------------------------------------------------------
__global__ __launch_bounds__(128) void k_proj(
    const bf16* __restrict__ a, const float* __restrict__ w,
    const float* __restrict__ bias, float* __restrict__ out)
{
  __shared__ alignas(16) bf16 As[64][32];
  __shared__ alignas(16) bf16 Bs[64][32];
  const int tid = threadIdx.x, lane = tid & 31, wv = tid >> 5;
  const int wm = wv >> 1, wn = wv & 1;
  const int m0 = blockIdx.x * 64, n0 = blockIdx.y * 64;
  const int half = lane >> 4, lr = lane & 15;

  v8f acc[2][2] = {};
  for (int kt = 0; kt < 512; kt += 32) {
    if (kt + 32 < 512) {
      const size_t pro = (size_t)(tid & 63) * 512 + kt + 32 + ((tid >> 6) << 4);
      __builtin_prefetch(w + (size_t)n0 * 512 + pro, 0, 0);
    }
    __syncthreads();
#if HAVE_ASYNC_LDS
    // A tile: 64 rows x 64B = 256 x b128 chunks, 2 per thread, DMA to LDS.
    for (int i = 0; i < 2; ++i) {
      int idx = tid + i * 128;
      int row = idx >> 2, c8 = (idx & 3) << 3;      // 8 bf16 = 16B chunks
      __builtin_amdgcn_global_load_async_to_lds_b128(
          (as1_v4i)(a + (size_t)(m0 + row) * 512 + kt + c8),
          (as3_v4i)&As[row][c8], 0, 0);
    }
#endif
    for (int i = 0; i < 4; ++i) {
      int idx = tid + i * 128;
      int row = idx >> 3, c4 = (idx & 7) << 2;
#if !HAVE_ASYNC_LDS
      uint2 av = *(const uint2*)(a + (size_t)(m0 + row) * 512 + kt + c4);
      *(uint2*)&As[row][c4] = av;
#endif
      float4 fb = *(const float4*)(w + (size_t)(n0 + row) * 512 + kt + c4);
      Bs[row][c4+0]=f2bf(fb.x); Bs[row][c4+1]=f2bf(fb.y);
      Bs[row][c4+2]=f2bf(fb.z); Bs[row][c4+3]=f2bf(fb.w);
    }
#if HAVE_ASYNC_LDS
    wait_asynccnt0();
#endif
    __syncthreads();
    v16bf af[2], bfr[2];
    for (int mt = 0; mt < 2; ++mt) {
      const bf16* p = &As[wm*32 + mt*16 + lr][half*8];
      V16 u; u.h[0] = *(const v8bf*)p; u.h[1] = *(const v8bf*)(p + 16);
      af[mt] = u.v;
    }
    for (int nt = 0; nt < 2; ++nt) {
      const bf16* p = &Bs[wn*32 + nt*16 + lr][half*8];
      V16 u; u.h[0] = *(const v8bf*)p; u.h[1] = *(const v8bf*)(p + 16);
      bfr[nt] = u.v;
    }
    for (int mt = 0; mt < 2; ++mt)
      for (int nt = 0; nt < 2; ++nt)
        acc[mt][nt] = __builtin_amdgcn_wmma_f32_16x16x32_bf16(
            false, af[mt], false, bfr[nt], (short)0, acc[mt][nt], false, false);
  }
  for (int mt = 0; mt < 2; ++mt)
    for (int nt = 0; nt < 2; ++nt) {
      int n = n0 + wn*32 + nt*16 + lr;
      float bi = bias[n];
      for (int r = 0; r < 8; ++r) {
        int row = m0 + wm*32 + mt*16 + r + 8*half;
        out[(size_t)row * 512 + n] = acc[mt][nt][r] + bi;
      }
    }
}

extern "C" void kernel_launch(void* const* d_in, const int* in_sizes, int n_in,
                              void* d_out, int out_size, void* d_ws, size_t ws_size,
                              hipStream_t stream) {
  const float* x      = (const float*)d_in[0];
  const float* w_qkv  = (const float*)d_in[1];
  const float* b_qkv  = (const float*)d_in[2];
  const float* rpb    = (const float*)d_in[3];
  const float* w_proj = (const float*)d_in[4];
  const float* b_proj = (const float*)d_in[5];
  float* out = (float*)d_out;

  // Workspace: q,k,v in (B,H,L,D) bf16 + attn_out (B,L,C) bf16 = 16 MB total.
  const size_t per = (size_t)Bc * Hc * Lc * Dc;       // 2,097,152 elems
  bf16* qb  = (bf16*)d_ws;
  bf16* kbf = qb  + per;
  bf16* vbf = kbf + per;
  bf16* ao  = vbf + per;                              // B*L*C elems

  k_qkv <<<dim3(64, 24), 128, 0, stream>>>(x, w_qkv, b_qkv, qb, kbf, vbf);
  k_natt<<<512,          128, 0, stream>>>(qb, kbf, vbf, rpb, ao);
  k_proj<<<dim3(64, 8),  128, 0, stream>>>(ao, w_proj, b_proj, out);
}
